// MultiHeadAttention_5059471475068
// MI455X (gfx1250) — matrix operand order repaired and checked by reading, not compiled, not run on hardware
//
#include <hip/hip_runtime.h>
#include <math.h>

// ---------------------------------------------------------------------------
// MI455X (gfx1250) multi-head attention forward, all matmuls on WMMA f16.
//   q = x Wq^T + bq ; k = ...; v = ...
//   S = K Q^T * 1/sqrt(DIM); P = softmax(S); O = P V ; out = O Wo^T + bo
// wave32 only; WMMA shape 16x16x32 (A:16x32 f16, B:32x16 f16, C/D:16x16 f32)
// ---------------------------------------------------------------------------

typedef _Float16 v16h __attribute__((ext_vector_type(16)));
typedef float    v8f  __attribute__((ext_vector_type(8)));
typedef _Float16 h2   __attribute__((ext_vector_type(2)));

#define DIMSZ 1024
#define HEADS 16
#define HDIM  64
#define BATCH 2
#define SEQ   2048

__device__ __forceinline__ v8f zero_v8f() {
    v8f z;
#pragma unroll
    for (int i = 0; i < 8; ++i) z[i] = 0.0f;
    return z;
}

// K index pattern of the f16 A-fragment (16x32), contiguous order:
// lanes 0-15: K = {0..15}; lanes 16-31: K = {16..31}
__device__ __forceinline__ int a_kidx(int i, int half) {
    return 2 * i + 16 * half;
}

// ---------------------------------------------------------------------------
// fp32 -> f16 conversion, 4 elements per lane (all sizes divisible by 1024)
// ---------------------------------------------------------------------------
__global__ void __launch_bounds__(256)
cvt_f32_f16(const float* __restrict__ src, _Float16* __restrict__ dst, int n) {
    int i = (blockIdx.x * blockDim.x + threadIdx.x) * 4;
    if (i + 3 < n) {
        float4 v = *(const float4*)(src + i);
        dst[i + 0] = (_Float16)v.x;
        dst[i + 1] = (_Float16)v.y;
        dst[i + 2] = (_Float16)v.z;
        dst[i + 3] = (_Float16)v.w;
    }
}

// ---------------------------------------------------------------------------
// C[M,N] = A[M,K] * W[N,K]^T + bias.  A,W f16 row-major.
// MODE 0: store f32 row-major [M,N]        (final output)
// MODE 1: store f16 row-major [M,N]        (Q, K, attention result)
// MODE 2: store f16 transposed [B,DIM,SEQ] (V: makes P.V B-fragments
//         contiguous 32-byte loads in the attention kernel)
// Block = 128 threads = 4 waves; wave computes a 16x64 C tile.
// grid = (M/64, N/64)
// ---------------------------------------------------------------------------
template <int MODE>
__global__ void __launch_bounds__(128, 1)
gemm_wmma(const _Float16* __restrict__ A,
          const _Float16* __restrict__ W,
          const float* __restrict__ bias,
          void* __restrict__ Cout,
          int M, int K, int Nout) {
    const int lane = threadIdx.x & 31;
    const int wave = threadIdx.x >> 5;
    const int half = lane >> 4;
    const int l16  = lane & 15;
    const int row0 = blockIdx.x * 64 + wave * 16;
    const int col0 = blockIdx.y * 64;
    const int arow = row0 + l16;

    v8f acc[4];
#pragma unroll
    for (int t = 0; t < 4; ++t) acc[t] = zero_v8f();

    for (int k0 = 0; k0 < K; k0 += 32) {
        // A fragment: lane holds row (lane%16), 16 f16 of the 32-wide K slice
        v16h afr;
#pragma unroll
        for (int i = 0; i < 8; ++i) {
            int kk = k0 + a_kidx(i, half);
            h2 p = *(const h2*)(A + (size_t)arow * K + kk);
            afr[2 * i + 0] = p.x;
            afr[2 * i + 1] = p.y;
        }
        if (k0 + 32 < K)  // hint next K-slice of this row into cache
            __builtin_prefetch(A + (size_t)arow * K + k0 + 32, 0, 0);

#pragma unroll
        for (int t = 0; t < 4; ++t) {
            // B fragment: lane holds col (lane%16); VGPR j -> K = 2j+16*half
            const int col = col0 + t * 16 + l16;
            v16h bfr;
#pragma unroll
            for (int j = 0; j < 8; ++j) {
                int kk = k0 + 2 * j + 16 * half;
                h2 p = *(const h2*)(W + (size_t)col * K + kk);
                bfr[2 * j + 0] = p.x;
                bfr[2 * j + 1] = p.y;
            }
            acc[t] = __builtin_amdgcn_wmma_f32_16x16x32_f16(
                false, afr, false, bfr, (short)0, acc[t], false, false);
        }
    }

#pragma unroll
    for (int t = 0; t < 4; ++t) {
        const int col = col0 + t * 16 + l16;
        const float bv = bias[col];
#pragma unroll
        for (int r = 0; r < 8; ++r) {
            const int row = row0 + r + 8 * half;
            float v = acc[t][r] + bv;
            if (MODE == 0) {
                ((float*)Cout)[(size_t)row * Nout + col] = v;
            } else if (MODE == 1) {
                ((_Float16*)Cout)[(size_t)row * Nout + col] = (_Float16)v;
            } else {  // MODE 2: [B, DIM, SEQ]
                const int bb = row / SEQ;
                const int n  = row % SEQ;
                ((_Float16*)Cout)[((size_t)bb * DIMSZ + col) * SEQ + n] =
                    (_Float16)v;
            }
        }
    }
}

// ---------------------------------------------------------------------------
// Flash attention, per (b,h). Q/K stored f16 [B,N,DIM] (head offset h*64);
// V stored f16 transposed [B,DIM,SEQ].
// scores[n,m] = sum_d K[n,d]*Q[m,d] * scale ; softmax over m ; O = P V.
// Block = 128 threads = 4 waves; wave owns 16 'n' rows, streams m in 32-chunks.
//
// Softmax: raw f32 scores bounce through LDS from C/D layout to A layout (the
// relayout is needed anyway for the P.V WMMA).  In A layout lane l owns row
// l%16, so row max / row sum are in-register + ONE shfl_xor(16) each; running
// stats live per-lane in the A domain; only the O-rescale factor is broadcast
// back to the C/D row domain (8 shfl per chunk).
// V fragments: with Vt[b][col][n] the 16 f16 of a B-fragment are contiguous,
// so each of the 4 tiles is a single 32B-aligned v16h load (2x b128).
// grid = (N/64, B*H)
// ---------------------------------------------------------------------------
__global__ void __launch_bounds__(128, 1)
attn_wmma(const _Float16* __restrict__ Qh,
          const _Float16* __restrict__ Kh,
          const _Float16* __restrict__ Vt,
          _Float16* __restrict__ Oh,
          int Nn, int Hh, float scale) {
    __shared__ float ldsS[4][16 * 32];  // per-wave 16x32 f32 S tile (2 KB each)

    const int lane = threadIdx.x & 31;
    const int wave = threadIdx.x >> 5;
    const int half = lane >> 4;
    const int l16  = lane & 15;
    const int bh = blockIdx.y;
    const int b  = bh / Hh;
    const int h  = bh % Hh;
    const int n0 = blockIdx.x * 64 + wave * 16;
    const size_t base  = (size_t)b * Nn * DIMSZ + (size_t)h * HDIM;
    // Vt base for this lane: rows col = h*64 + dc, dc = t*16+l16; n contiguous
    const size_t baseV = ((size_t)b * DIMSZ + (size_t)h * HDIM + l16) * SEQ +
                         16 * half;

    // A fragments of K rows [n0..n0+15] x d[0..63]  (two 32-wide K-steps)
    v16h aK[2];
#pragma unroll
    for (int ks = 0; ks < 2; ++ks)
#pragma unroll
        for (int i = 0; i < 8; ++i) {
            int dd = 32 * ks + a_kidx(i, half);
            h2 p = *(const h2*)(Kh + base + (size_t)(n0 + l16) * DIMSZ + dd);
            aK[ks][2 * i + 0] = p.x;
            aK[ks][2 * i + 1] = p.y;
        }

    v8f Oacc[4];
#pragma unroll
    for (int t = 0; t < 4; ++t) Oacc[t] = zero_v8f();
    // per-lane running stats for row l16 (duplicated in both 16-lane halves)
    float mrun = -3.0e38f;
    float lrun = 0.0f;

    for (int m0 = 0; m0 < Nn; m0 += 32) {
        // ---- S = K . Q^T over this 32-wide m chunk (two 16-col tiles) ----
        v8f s[2];
#pragma unroll
        for (int t = 0; t < 2; ++t) {
            const int m = m0 + t * 16 + l16;  // B-fragment col
            v16h bq0, bq1;
#pragma unroll
            for (int j = 0; j < 8; ++j) {
                int d0 = 2 * j + 16 * half;
                h2 p0 = *(const h2*)(Qh + base + (size_t)m * DIMSZ + d0);
                h2 p1 = *(const h2*)(Qh + base + (size_t)m * DIMSZ + d0 + 32);
                bq0[2 * j] = p0.x; bq0[2 * j + 1] = p0.y;
                bq1[2 * j] = p1.x; bq1[2 * j + 1] = p1.y;
            }
            v8f z = zero_v8f();
            z = __builtin_amdgcn_wmma_f32_16x16x32_f16(false, aK[0], false, bq0,
                                                       (short)0, z, false, false);
            z = __builtin_amdgcn_wmma_f32_16x16x32_f16(false, aK[1], false, bq1,
                                                       (short)0, z, false, false);
#pragma unroll
            for (int r = 0; r < 8; ++r) z[r] *= scale;
            s[t] = z;
        }

        // ---- bounce raw f32 scores C/D layout -> A layout via LDS ----
#pragma unroll
        for (int t = 0; t < 2; ++t)
#pragma unroll
            for (int r = 0; r < 8; ++r)
                ldsS[wave][(r + 8 * half) * 32 + t * 16 + l16] = s[t][r];
        // cross-lane LDS dependency the compiler cannot see: drain DScnt
        asm volatile("s_wait_dscnt 0x0" ::: "memory");

        // lane owns row l16; its 16 of 32 cols follow the A-fragment pattern
        float rv[16];
        float mx = -3.0e38f;
#pragma unroll
        for (int i = 0; i < 8; ++i) {
            int kk = a_kidx(i, half);
            float2 p = *(const float2*)&ldsS[wave][l16 * 32 + kk];
            rv[2 * i + 0] = p.x;
            rv[2 * i + 1] = p.y;
            mx = fmaxf(mx, fmaxf(p.x, p.y));
        }
        mx = fmaxf(mx, __shfl_xor(mx, 16));  // merge the two half-rows
        const float mnew = fmaxf(mrun, mx);
        const float cf = __expf(mrun - mnew);

        // exp in-register, pack straight into the P A-fragment, local row sum
        v16h aP;
        float sm = 0.0f;
#pragma unroll
        for (int e = 0; e < 16; ++e) {
            float p = __expf(rv[e] - mnew);
            sm += p;
            aP[e] = (_Float16)p;
        }
        sm += __shfl_xor(sm, 16);
        lrun = lrun * cf + sm;
        mrun = mnew;

        // broadcast rescale factor to the C/D row domain and rescale O
        float cfb[8];
#pragma unroll
        for (int r = 0; r < 8; ++r) cfb[r] = __shfl(cf, r + 8 * half);
#pragma unroll
        for (int t = 0; t < 4; ++t)
#pragma unroll
            for (int r = 0; r < 8; ++r) Oacc[t][r] *= cfb[r];

        // ---- O += P . V : each V B-fragment is one contiguous v16h load ----
#pragma unroll
        for (int t = 0; t < 4; ++t) {
            v16h bv = *(const v16h*)(Vt + baseV + (size_t)t * 16 * SEQ + m0);
            Oacc[t] = __builtin_amdgcn_wmma_f32_16x16x32_f16(
                false, aP, false, bv, (short)0, Oacc[t], false, false);
        }
    }

    // ---- finalize: divide by row sum (broadcast once), write f16 ----
    float lb[8];
#pragma unroll
    for (int r = 0; r < 8; ++r) lb[r] = __shfl(lrun, r + 8 * half);
#pragma unroll
    for (int t = 0; t < 4; ++t) {
        const int dc = t * 16 + l16;
#pragma unroll
        for (int r = 0; r < 8; ++r) {
            const int n = n0 + r + 8 * half;
            float o = Oacc[t][r] / lb[r];
            Oh[base + (size_t)n * DIMSZ + dc] = (_Float16)o;
        }
    }
}

// ---------------------------------------------------------------------------
// Launch: d_in = {x, Wq, bq, Wk, bk, Wv, bv, Wo, bo}, all fp32.
// Workspace layout (f16):
//   [0,8MB)   xh, later reused as attention output (atth)
//   [8,16MB)  Wq,Wk,Wv,Wo (2MB each)
//   [16,32MB) q,k (8MB each)
//   [32,40MB) vt = V transposed [B,DIM,SEQ]          total: 40 MB of d_ws
// ---------------------------------------------------------------------------
extern "C" void kernel_launch(void* const* d_in, const int* in_sizes, int n_in,
                              void* d_out, int out_size, void* d_ws,
                              size_t ws_size, hipStream_t stream) {
    (void)in_sizes; (void)n_in; (void)out_size; (void)ws_size;

    const float* x  = (const float*)d_in[0];
    const float* Wq = (const float*)d_in[1];
    const float* bq = (const float*)d_in[2];
    const float* Wk = (const float*)d_in[3];
    const float* bk = (const float*)d_in[4];
    const float* Wv = (const float*)d_in[5];
    const float* bv = (const float*)d_in[6];
    const float* Wo = (const float*)d_in[7];
    const float* bo = (const float*)d_in[8];

    const int M = BATCH * SEQ;          // 4096
    const int K = DIMSZ;                // 1024
    const int Nw = DIMSZ;               // 1024
    const size_t MB = 1024 * 1024;

    char* ws = (char*)d_ws;
    _Float16* xh   = (_Float16*)(ws + 0);        // also atth after attention
    _Float16* wqh  = (_Float16*)(ws + 8 * MB);
    _Float16* wkh  = (_Float16*)(ws + 10 * MB);
    _Float16* wvh  = (_Float16*)(ws + 12 * MB);
    _Float16* woh  = (_Float16*)(ws + 14 * MB);
    _Float16* qh   = (_Float16*)(ws + 16 * MB);
    _Float16* kh   = (_Float16*)(ws + 24 * MB);
    _Float16* vt   = (_Float16*)(ws + 32 * MB);  // [B, DIM, SEQ]

    const int nx = M * DIMSZ;           // 4,194,304
    const int nw = DIMSZ * DIMSZ;       // 1,048,576
    cvt_f32_f16<<<nx / 1024, 256, 0, stream>>>(x, xh, nx);
    cvt_f32_f16<<<nw / 1024, 256, 0, stream>>>(Wq, wqh, nw);
    cvt_f32_f16<<<nw / 1024, 256, 0, stream>>>(Wk, wkh, nw);
    cvt_f32_f16<<<nw / 1024, 256, 0, stream>>>(Wv, wvh, nw);
    cvt_f32_f16<<<nw / 1024, 256, 0, stream>>>(Wo, woh, nw);

    dim3 ggrid(M / 64, Nw / 64);        // (64,16)
    gemm_wmma<1><<<ggrid, 128, 0, stream>>>(xh, wqh, bq, (void*)qh, M, K, Nw);
    gemm_wmma<1><<<ggrid, 128, 0, stream>>>(xh, wkh, bk, (void*)kh, M, K, Nw);
    gemm_wmma<2><<<ggrid, 128, 0, stream>>>(xh, wvh, bv, (void*)vt, M, K, Nw);

    const float scale = 1.0f / sqrtf((float)DIMSZ);
    dim3 agrid(SEQ / 64, BATCH * HEADS);  // (32,32)
    attn_wmma<<<agrid, 128, 0, stream>>>(qh, kh, vt, /*atth=*/xh, SEQ, HEADS,
                                         scale);

    gemm_wmma<0><<<ggrid, 128, 0, stream>>>(/*atth=*/xh, woh, bo, d_out, M,
                                            K, Nw);
}
